// KHopSumAggregator_9801115369800
// MI455X (gfx1250) — matrix operand order repaired and checked by reading, not compiled, not run on hardware
//
#include <hip/hip_runtime.h>

#define N_NODES 4096
#define NWORDS  128        // 4096 bits / 32
#define BATCH   4
#define FDIM    64
#define KHOP    3
#define MPOW    4

#define LDS_FSTRIDE 40           // halves per f-row (80 B, pads 64 banks conflict-free)
#define BUFH (FDIM * LDS_FSTRIDE)       // halves per buffer = 2560 (5120 B)

typedef __attribute__((ext_vector_type(16))) _Float16 v16h;
typedef __attribute__((ext_vector_type(8)))  float    v8f;

// ---------------------------------------------------------------- utilities
__global__ void zero_words_kernel(unsigned* __restrict__ p, int n) {
    int i = blockIdx.x * blockDim.x + threadIdx.x;
    if (i < n) p[i] = 0u;
}

// adj[t][s] bitset: bit s of row t set iff edge s->t exists (A[t,s]=1)
__global__ void build_adj_kernel(const int* __restrict__ ei, int E,
                                 unsigned* __restrict__ adj) {
    int e = blockIdx.x * blockDim.x + threadIdx.x;
    if (e >= E) return;
    int s = ei[e];          // edge_index[0, e] (src)
    int t = ei[E + e];      // edge_index[1, e] (dst)
    atomicOr(&adj[(size_t)t * NWORDS + (s >> 5)], 1u << (s & 31));
}

// reach1 = adj | I
__global__ void init_reach1_kernel(const unsigned* __restrict__ adj,
                                   unsigned* __restrict__ r1) {
    int idx = blockIdx.x * blockDim.x + threadIdx.x;   // 4096*128 words
    int i = idx >> 7, w = idx & (NWORDS - 1);
    unsigned v = adj[idx];
    if (w == (i >> 5)) v |= 1u << (i & 31);
    r1[idx] = v;
}

// next[i] = prev[i] | OR_{j set in prev[i]} adj[j]
__global__ void expand_reach_kernel(const unsigned* __restrict__ prev,
                                    const unsigned* __restrict__ adj,
                                    unsigned* __restrict__ next) {
    __shared__ unsigned rowbits[NWORDS];
    const int i = blockIdx.x;
    const int w = threadIdx.x;
    unsigned p = prev[(size_t)i * NWORDS + w];
    rowbits[w] = p;
    __syncthreads();
    unsigned acc = p;
    for (int w2 = 0; w2 < NWORDS; ++w2) {
        unsigned bits = rowbits[w2];
        while (bits) {
            int b = __ffs(bits) - 1;
            bits &= bits - 1u;
            int j = (w2 << 5) + b;
            acc |= adj[(size_t)j * NWORDS + w];   // coalesced across w
        }
    }
    next[(size_t)i * NWORDS + w] = acc;
}

// ph[((b*MPOW+m)*FDIM+f)*N + j] = |x[b,j,f]|^(m+1)  (f16, j contiguous)
__global__ void build_pows_kernel(const float* __restrict__ x,
                                  _Float16* __restrict__ ph) {
    int idx = blockIdx.x * blockDim.x + threadIdx.x;   // B*N*F, j fastest
    int j = idx & (N_NODES - 1);
    int f = (idx >> 12) & (FDIM - 1);
    int b = idx >> 18;
    float a  = fabsf(x[((size_t)b * N_NODES + j) * FDIM + f]);
    float a2 = a * a;
    size_t base = ((size_t)(b * MPOW) * FDIM + f) * N_NODES + j;
    const size_t ms = (size_t)FDIM * N_NODES;
    ph[base]          = (_Float16)a;
    ph[base + ms]     = (_Float16)a2;
    ph[base + 2 * ms] = (_Float16)(a2 * a);
    ph[base + 3 * ms] = (_Float16)(a2 * a2);
}

// ---------------------------------------------------- CDNA5 async helpers
__device__ __forceinline__ void async_copy_b128(unsigned lds_byte_off,
                                                unsigned long long gaddr) {
    // GLOBAL_LOAD_ASYNC_TO_LDS_B128 (GV mode), tracked by ASYNCcnt
    asm volatile("global_load_async_to_lds_b128 %0, %1, off"
                 :: "v"(lds_byte_off), "v"(gaddr) : "memory");
}
__device__ __forceinline__ void wait_async0() {
    asm volatile("s_wait_asynccnt 0x0" ::: "memory");
}

__device__ __forceinline__ unsigned u4sel(uint4 v, int u) {
    switch (u) { case 0: return v.x; case 1: return v.y;
                 case 2: return v.z; default: return v.w; }
}

// ------------------------------------------------------------- WMMA GEMM
// Workgroup = 8 waves, fixed (b,m): wave w owns i-tile (8*blk + w), ALL 3 hops.
// B panel (32 j x 64 f, f16, 4KB) async-staged to LDS once per WG per j-tile,
// double buffered -> 8 waves x 3 hops = 24 WMMA-uses per staged tile.
__global__ void __launch_bounds__(256)
khop_wmma_kernel(const unsigned* __restrict__ reach,   // [3][4096][128]
                 const _Float16* __restrict__ ph,      // [B*M][F][N]
                 float* __restrict__ out) {            // [B][N][K][M][F]
    __shared__ __align__(16) _Float16 smem[2 * BUFH];

    const int lane  = threadIdx.x & 31;
    const int wave  = threadIdx.x >> 5;                // 0..7
    const int itg   = blockIdx.x & 31;                 // i-tile group
    const int bm    = blockIdx.x >> 5;                 // 0..15
    const int m     = bm & (MPOW - 1);
    const int b     = bm >> 2;
    const int itile = itg * 8 + wave;

    const int col    = lane & 15;
    const int kbase0 = (lane & 16) ? 16 : 0;           // A-frag K offsets
    const int jsh    = (lane & 16) ? 16 : 0;           // B-frag K offset

    const _Float16* __restrict__ panel =
        ph + (size_t)(b * MPOW + m) * FDIM * N_NODES;

    // reach row pointers (one per hop), read as uint4 = 4 j-tiles per load
    const int row = itile * 16 + col;
    const uint4* __restrict__ rq0 = (const uint4*)(reach + ((size_t)0 * N_NODES + row) * NWORDS);
    const uint4* __restrict__ rq1 = (const uint4*)(reach + ((size_t)1 * N_NODES + row) * NWORDS);
    const uint4* __restrict__ rq2 = (const uint4*)(reach + ((size_t)2 * N_NODES + row) * NWORDS);

    // cooperative copy mapping: thread t moves 16B; f = t/4, j-block = (t%4)*8
    const int ct = threadIdx.x;
    const int cf = ct >> 2;
    const int cj = (ct & 3) * 8;
    const unsigned smem_off = (unsigned)(size_t)(&smem[0]);
    const unsigned lds_copy = smem_off + (unsigned)((cf * LDS_FSTRIDE + cj) * 2);
    const unsigned long long g_copy =
        (unsigned long long)(size_t)(panel + (size_t)cf * N_NODES + cj);

    v8f acc[KHOP][4] = {};

    // prologue: stage j-tile 0 into buffer 0
    async_copy_b128(lds_copy, g_copy);
    wait_async0();
    __syncthreads();

    const int fragoff = col * LDS_FSTRIDE + jsh;

    for (int q = 0; q < NWORDS / 4; ++q) {
        uint4 w0 = rq0[q];
        uint4 w1 = rq1[q];
        uint4 w2 = rq2[q];
#pragma unroll
        for (int u = 0; u < 4; ++u) {
            const int jt  = q * 4 + u;
            const int buf = jt & 1;

            // stage j-tile jt+1 into the other buffer (overlaps WMMAs below)
            if (jt + 1 < NWORDS)
                async_copy_b128(lds_copy + (unsigned)((buf ^ 1) * BUFH * 2),
                                g_copy + (unsigned long long)(jt + 1) * 64u);

            // B fragments from LDS (two ds_load_b128 each, bank-conflict-free)
            const _Float16* sb = smem + buf * BUFH + fragoff;
            v16h bf0 = *(const v16h*)(sb);
            v16h bf1 = *(const v16h*)(sb + 16 * LDS_FSTRIDE);
            v16h bf2 = *(const v16h*)(sb + 32 * LDS_FSTRIDE);
            v16h bf3 = *(const v16h*)(sb + 48 * LDS_FSTRIDE);

#pragma unroll
            for (int kk = 0; kk < KHOP; ++kk) {
                unsigned word = u4sel(kk == 0 ? w0 : (kk == 1 ? w1 : w2), u);
                v16h a;
#pragma unroll
                for (int e = 0; e < 8; ++e) {
                    a[e]     = ((word >> (kbase0 + e))      & 1u) ? (_Float16)1.0f
                                                                  : (_Float16)0.0f;
                    a[e + 8] = ((word >> (kbase0 + 8 + e)) & 1u) ? (_Float16)1.0f
                                                                  : (_Float16)0.0f;
                }
                acc[kk][0] = __builtin_amdgcn_wmma_f32_16x16x32_f16(
                    false, a, false, bf0, (short)0, acc[kk][0], false, false);
                acc[kk][1] = __builtin_amdgcn_wmma_f32_16x16x32_f16(
                    false, a, false, bf1, (short)0, acc[kk][1], false, false);
                acc[kk][2] = __builtin_amdgcn_wmma_f32_16x16x32_f16(
                    false, a, false, bf2, (short)0, acc[kk][2], false, false);
                acc[kk][3] = __builtin_amdgcn_wmma_f32_16x16x32_f16(
                    false, a, false, bf3, (short)0, acc[kk][3], false, false);
            }

            wait_async0();      // next tile landed
            __syncthreads();    // all waves done reading current buffer
        }
    }

    // D layout: elem r -> row i0 + r + 8*(lane>=16), col = lane&15
    const int ib = itile * 16 + ((lane & 16) ? 8 : 0);
#pragma unroll
    for (int kk = 0; kk < KHOP; ++kk) {
#pragma unroll
        for (int r = 0; r < 8; ++r) {
            size_t o = ((((size_t)b * N_NODES + (ib + r)) * KHOP + kk) * MPOW + m)
                       * FDIM + col;
            out[o +  0] = acc[kk][0][r];
            out[o + 16] = acc[kk][1][r];
            out[o + 32] = acc[kk][2][r];
            out[o + 48] = acc[kk][3][r];
        }
    }
}

// ---------------------------------------------------------------- launcher
extern "C" void kernel_launch(void* const* d_in, const int* in_sizes, int n_in,
                              void* d_out, int out_size, void* d_ws, size_t ws_size,
                              hipStream_t stream) {
    const float* x  = (const float*)d_in[0];
    const int*   ei = (const int*)d_in[1];
    float*       out = (float*)d_out;
    const int E = in_sizes[1] / 2;

    unsigned char* ws = (unsigned char*)d_ws;
    unsigned* adj   = (unsigned*)ws;                                  // 2 MB
    unsigned* reach = (unsigned*)(ws + ((size_t)2 << 20));            // 6 MB
    _Float16* ph    = (_Float16*)(ws + ((size_t)8 << 20));            // 8 MB

    const int totalWords = N_NODES * NWORDS;                          // 524288

    zero_words_kernel<<<totalWords / 256, 256, 0, stream>>>(adj, totalWords);
    build_adj_kernel<<<(E + 255) / 256, 256, 0, stream>>>(ei, E, adj);
    init_reach1_kernel<<<totalWords / 256, 256, 0, stream>>>(adj, reach);
    expand_reach_kernel<<<N_NODES, NWORDS, 0, stream>>>(
        reach, adj, reach + (size_t)totalWords);
    expand_reach_kernel<<<N_NODES, NWORDS, 0, stream>>>(
        reach + (size_t)totalWords, adj, reach + (size_t)2 * totalWords);

    build_pows_kernel<<<(BATCH * N_NODES * FDIM) / 256, 256, 0, stream>>>(x, ph);

    // 16 (b,m) combos x 32 i-tile groups; 8 waves per WG (one i-tile each)
    khop_wmma_kernel<<<BATCH * MPOW * 32, 256, 0, stream>>>(reach, ph, out);
}
